// Net_16114717294603
// MI455X (gfx1250) — compile-verified, matrix operand order checked by reading, not run on hardware
//
#include <hip/hip_runtime.h>
#include <hip/hip_bf16.h>
#include <math.h>

typedef __attribute__((ext_vector_type(16))) _Float16 v16h;
typedef __attribute__((ext_vector_type(8)))  float    v8f;

#define D 128
#define NB 64

// ---------------------------------------------------------------------------
// WMMA GEMM: Y[m][n] = sum_k X[m][k] * W[n][k] + bias[n]
// X: [M x 128] f32, W: [128 x 128] f32 (row-major, rows are output features)
// Each wave computes a 16x16 f32 tile via v_wmma_f32_16x16x32_f16 (K=128 in 4 steps).
// Block = 256 threads = 8 waves arranged 4 (rows) x 2 (cols) -> 64x32 per block.
// ---------------------------------------------------------------------------
__device__ __forceinline__ v16h load_frag_f32_to_f16(const float* p) {
    // p points at row_base + k0 + kb ; elements 0..7 <- K=kb..kb+7, 8..15 <- K=16+kb..
    const float4 a0 = *(const float4*)(p);
    const float4 a1 = *(const float4*)(p + 4);
    const float4 a2 = *(const float4*)(p + 16);
    const float4 a3 = *(const float4*)(p + 20);
    v16h f;
    f[0]=(_Float16)a0.x;  f[1]=(_Float16)a0.y;  f[2]=(_Float16)a0.z;  f[3]=(_Float16)a0.w;
    f[4]=(_Float16)a1.x;  f[5]=(_Float16)a1.y;  f[6]=(_Float16)a1.z;  f[7]=(_Float16)a1.w;
    f[8]=(_Float16)a2.x;  f[9]=(_Float16)a2.y;  f[10]=(_Float16)a2.z; f[11]=(_Float16)a2.w;
    f[12]=(_Float16)a3.x; f[13]=(_Float16)a3.y; f[14]=(_Float16)a3.z; f[15]=(_Float16)a3.w;
    return f;
}

__global__ void gemm_xwT_wmma(const float* __restrict__ X, const float* __restrict__ W,
                              const float* __restrict__ bias, float* __restrict__ Y, int M) {
    const int lane = threadIdx.x & 31;
    const int wave = threadIdx.x >> 5;       // 0..7
    const int wr   = wave & 3;               // row-tile within block (0..3)
    const int wc   = wave >> 2;              // col-tile within block (0..1)
    const int rbase = blockIdx.x * 64 + wr * 16;
    const int cbase = blockIdx.y * 32 + wc * 16;
    const int half  = lane >> 4;             // 0 or 1
    const int l16   = lane & 15;
    const int kb    = half * 8;
    const int arow  = rbase + l16;           // A-matrix row (M index)
    const int brow  = cbase + l16;           // W row == output column (N index)
    const bool apred = (arow < M);

    v8f acc; for (int r = 0; r < 8; ++r) acc[r] = 0.0f;

    for (int k0 = 0; k0 < D; k0 += 32) {
        v16h af, bf;
        if (apred) {
            af = load_frag_f32_to_f16(X + (size_t)arow * D + k0 + kb);
        } else {
            for (int j = 0; j < 16; ++j) af[j] = (_Float16)0.0f;
        }
        bf = load_frag_f32_to_f16(W + (size_t)brow * D + k0 + kb);
        acc = __builtin_amdgcn_wmma_f32_16x16x32_f16(
                  /*neg_a=*/false, af, /*neg_b=*/false, bf,
                  /*c_mod=*/(short)0, acc, /*reuse_a=*/false, /*reuse_b=*/false);
    }

    const int col = cbase + l16;             // D-matrix: N = lane%16
    const float bv = bias[col];
    for (int r = 0; r < 8; ++r) {
        const int m = rbase + half * 8 + r;  // D-matrix: M = r + 8*(lane>=16)
        if (m < M) Y[(size_t)m * D + col] = acc[r] + bv;
    }
}

// ---------------------------------------------------------------------------
// Graph kernels
// ---------------------------------------------------------------------------
__global__ void init_valid_kernel(float* __restrict__ valid, int E) {
    int e = blockIdx.x * blockDim.x + threadIdx.x;
    if (e < E) valid[e] = 1.0f;
}

__global__ void deg_kernel(const int* __restrict__ src, const float* __restrict__ valid,
                           float* __restrict__ deg, int E) {
    int e = blockIdx.x * blockDim.x + threadIdx.x;
    if (e < E) atomicAdd(&deg[src[e]], valid[e]);
}

__global__ void dis_kernel(const float* __restrict__ deg, float* __restrict__ dis, int M) {
    int n = blockIdx.x * blockDim.x + threadIdx.x;
    if (n < M) dis[n] = rsqrtf(deg[n] + 1.0f);
}

// one (edge, feature) pair per thread: agg[dst] += dis[s]*dis[d]*valid * Y[src]
__global__ void agg_kernel(const float* __restrict__ Y, const int* __restrict__ src,
                           const int* __restrict__ dst, const float* __restrict__ valid,
                           const float* __restrict__ dis, float* __restrict__ agg, int E) {
    int idx = blockIdx.x * blockDim.x + threadIdx.x;
    int e = idx >> 7, f = idx & 127;
    if (e >= E) return;
    float v = valid[e];
    if (v == 0.0f) return;
    int s = src[e], d = dst[e];
    float nrm = dis[s] * dis[d] * v;
    atomicAdd(&agg[(size_t)d * D + f], nrm * Y[(size_t)s * D + f]);
}

// in-place: agg = relu(agg + dis^2 * Y)
__global__ void self_relu_kernel(float* __restrict__ agg, const float* __restrict__ Y,
                                 const float* __restrict__ dis, int M) {
    int idx = blockIdx.x * blockDim.x + threadIdx.x;
    int n = idx >> 7;
    if (n >= M) return;
    float di = dis[n];
    float val = agg[idx] + di * di * Y[idx];
    agg[idx] = fmaxf(val, 0.0f);
}

__global__ void wnorm_kernel(const float* __restrict__ w, float* __restrict__ invn) {
    __shared__ float red[D];
    int t = threadIdx.x;
    float v = w[t];
    red[t] = v * v;
    __syncthreads();
    for (int s = D / 2; s > 0; s >>= 1) {
        if (t < s) red[t] += red[t + s];
        __syncthreads();
    }
    if (t == 0) *invn = rsqrtf(red[0]);
}

__global__ void score_kernel(const float* __restrict__ X, const float* __restrict__ w,
                             const float* __restrict__ invn, float* __restrict__ s, int M) {
    int n = blockIdx.x * blockDim.x + threadIdx.x;
    if (n >= M) return;
    float acc = 0.0f;
    const float* p = X + (size_t)n * D;
    for (int f = 0; f < D; ++f) acc += p[f] * w[f];
    s[n] = tanhf(acc * (*invn));
}

// per-graph bitonic top-k (block = 1024 threads, n <= 1024)
__global__ void topk_kernel(const float* __restrict__ score, int n, int k,
                            int* __restrict__ perm_old, float* __restrict__ scale,
                            int* __restrict__ new_id) {
    __shared__ float sv[1024];
    __shared__ int   si[1024];
    const int b = blockIdx.x, t = threadIdx.x;
    sv[t] = (t < n) ? score[b * n + t] : -INFINITY;
    si[t] = t;
    __syncthreads();
    for (int size = 2; size <= 1024; size <<= 1) {
        for (int stride = size >> 1; stride > 0; stride >>= 1) {
            int p = t ^ stride;
            if (p > t) {
                float va = sv[t], vb = sv[p];
                int   ia = si[t], ib = si[p];
                bool descending = ((t & size) == 0);
                bool doswap = descending ? (va < vb || (va == vb && ia > ib))
                                         : (va > vb || (va == vb && ia < ib));
                if (doswap) { sv[t] = vb; sv[p] = va; si[t] = ib; si[p] = ia; }
            }
            __syncthreads();
        }
    }
    if (t < k) {
        int g = b * n + si[t];
        perm_old[b * k + t] = g;
        scale[b * k + t]    = sv[t];
        new_id[g]           = b * k + t;
    }
}

__global__ void gather_kernel(const float* __restrict__ Xin, const int* __restrict__ perm_old,
                              const float* __restrict__ scale, float* __restrict__ Xout, int Nk) {
    int idx = blockIdx.x * blockDim.x + threadIdx.x;
    int j = idx >> 7, f = idx & 127;
    if (j >= Nk) return;
    Xout[idx] = Xin[(size_t)perm_old[j] * D + f] * scale[j];
}

__global__ void remap_kernel(int* __restrict__ src, int* __restrict__ dst,
                             float* __restrict__ valid, const int* __restrict__ new_id, int E) {
    int e = blockIdx.x * blockDim.x + threadIdx.x;
    if (e >= E) return;
    int ns = new_id[src[e]];
    int nd = new_id[dst[e]];
    float v = valid[e] * ((ns >= 0 && nd >= 0) ? 1.0f : 0.0f);
    src[e] = ns < 0 ? 0 : ns;
    dst[e] = nd < 0 ? 0 : nd;
    valid[e] = v;
}

// per-graph max+mean over k rows, accumulate into h[B,256]
__global__ void readout_kernel(const float* __restrict__ X, float* __restrict__ h, int k) {
    int b = blockIdx.x, f = threadIdx.x;   // blockDim = 128
    float mx = -INFINITY, sm = 0.0f;
    const float* p = X + (size_t)b * k * D + f;
    for (int j = 0; j < k; ++j) {
        float v = p[(size_t)j * D];
        mx = fmaxf(mx, v);
        sm += v;
    }
    h[b * 256 + f]       += mx;
    h[b * 256 + D + f]   += sm / (float)k;
}

// ---------------------------------------------------------------------------
// MLP head
// ---------------------------------------------------------------------------
__global__ void mlp_relu_kernel(const float* __restrict__ in, const float* __restrict__ Wt,
                                const float* __restrict__ bias, float* __restrict__ out, int Din) {
    int b = blockIdx.x, o = threadIdx.x;   // blockDim = Dout
    float acc = bias[o];
    const float* pi = in + (size_t)b * Din;
    const float* pw = Wt + (size_t)o * Din;
    for (int i = 0; i < Din; ++i) acc += pi[i] * pw[i];
    out[(size_t)b * blockDim.x + o] = fmaxf(acc, 0.0f);
}

__global__ void head_kernel(const float* __restrict__ h2, const float* __restrict__ l3w,
                            const float* __restrict__ l3b, float* __restrict__ out) {
    int b = blockIdx.x * blockDim.x + threadIdx.x;
    if (b >= NB) return;
    float z0 = l3b[0], z1 = l3b[1];
    const float* p = h2 + (size_t)b * 64;
    for (int i = 0; i < 64; ++i) {
        z0 += p[i] * l3w[i];
        z1 += p[i] * l3w[64 + i];
    }
    float m = fmaxf(z0, z1);
    float lse = m + logf(expf(z0 - m) + expf(z1 - m));
    out[b * 2 + 0] = z0 - lse;
    out[b * 2 + 1] = z1 - lse;
}

// ---------------------------------------------------------------------------
// Orchestration
// ---------------------------------------------------------------------------
extern "C" void kernel_launch(void* const* d_in, const int* in_sizes, int n_in,
                              void* d_out, int out_size, void* d_ws, size_t ws_size,
                              hipStream_t stream) {
    static const int NS[6] = {1024, 820, 656, 525, 420, 336};

    const float* x_in = (const float*)d_in[0];
    const int*   eidx = (const int*)d_in[1];
    const int    E    = in_sizes[1] / 2;     // 1,048,576

    const float* cw[5], *cb[5], *pw[5];
    for (int i = 0; i < 5; ++i) {
        cw[i] = (const float*)d_in[3 + 3 * i];
        cb[i] = (const float*)d_in[4 + 3 * i];
        pw[i] = (const float*)d_in[5 + 3 * i];
    }
    const float* l1w = (const float*)d_in[18];
    const float* l1b = (const float*)d_in[19];
    const float* l2w = (const float*)d_in[20];
    const float* l2b = (const float*)d_in[21];
    const float* l3w = (const float*)d_in[22];
    const float* l3b = (const float*)d_in[23];
    float* out = (float*)d_out;

    // scratch layout
    char* ws = (char*)d_ws;
    size_t off = 0;
    auto carve = [&](size_t bytes) { void* p = ws + off; off += (bytes + 255) & ~size_t(255); return p; };
    const size_t MAXN = (size_t)NB * 1024;
    float* xbuf   = (float*)carve(MAXN * D * 4);
    float* ybuf   = (float*)carve(MAXN * D * 4);
    float* aggbuf = (float*)carve(MAXN * D * 4);
    float* deg    = (float*)carve(MAXN * 4);
    float* dis    = (float*)carve(MAXN * 4);
    float* score  = (float*)carve(MAXN * 4);
    int*   perm   = (int*)  carve(MAXN * 4);
    float* scale  = (float*)carve(MAXN * 4);
    int*   newid  = (int*)  carve(MAXN * 4);
    int*   srcw   = (int*)  carve((size_t)E * 4);
    int*   dstw   = (int*)  carve((size_t)E * 4);
    float* validw = (float*)carve((size_t)E * 4);
    float* h      = (float*)carve((size_t)NB * 256 * 4);
    float* h1     = (float*)carve((size_t)NB * 128 * 4);
    float* h2     = (float*)carve((size_t)NB * 64 * 4);
    float* invn   = (float*)carve(256);

    // init edges (do not mutate inputs) + valid mask + readout accumulator
    hipMemcpyAsync(srcw, eidx,     (size_t)E * 4, hipMemcpyDeviceToDevice, stream);
    hipMemcpyAsync(dstw, eidx + E, (size_t)E * 4, hipMemcpyDeviceToDevice, stream);
    init_valid_kernel<<<(E + 255) / 256, 256, 0, stream>>>(validw, E);
    hipMemsetAsync(h, 0, (size_t)NB * 256 * 4, stream);

    const float* X = x_in;
    for (int layer = 0; layer < 5; ++layer) {
        const int n = NS[layer], k = NS[layer + 1];
        const int M = NB * n;

        // y = x @ W^T + b   (WMMA)
        dim3 gw((M + 63) / 64, 4);
        gemm_xwT_wmma<<<gw, 256, 0, stream>>>(X, cw[layer], cb[layer], ybuf, M);

        // degree + norm
        hipMemsetAsync(deg, 0, (size_t)M * 4, stream);
        deg_kernel<<<(E + 255) / 256, 256, 0, stream>>>(srcw, validw, deg, E);
        dis_kernel<<<(M + 255) / 256, 256, 0, stream>>>(deg, dis, M);

        // aggregation + self loop + relu
        hipMemsetAsync(aggbuf, 0, (size_t)M * D * 4, stream);
        agg_kernel<<<((size_t)E * D + 255) / 256, 256, 0, stream>>>(ybuf, srcw, dstw, validw, dis, aggbuf, E);
        self_relu_kernel<<<((size_t)M * D + 255) / 256, 256, 0, stream>>>(aggbuf, ybuf, dis, M);

        // top-k pooling
        wnorm_kernel<<<1, D, 0, stream>>>(pw[layer], invn);
        score_kernel<<<(M + 255) / 256, 256, 0, stream>>>(aggbuf, pw[layer], invn, score, M);
        hipMemsetAsync(newid, 0xFF, (size_t)M * 4, stream);
        topk_kernel<<<NB, 1024, 0, stream>>>(score, n, k, perm, scale, newid);
        gather_kernel<<<((size_t)NB * k * D + 255) / 256, 256, 0, stream>>>(aggbuf, perm, scale, xbuf, NB * k);
        remap_kernel<<<(E + 255) / 256, 256, 0, stream>>>(srcw, dstw, validw, newid, E);

        // readout (max || mean), accumulated across layers
        readout_kernel<<<NB, D, 0, stream>>>(xbuf, h, k);

        X = xbuf;
    }

    // MLP head + log_softmax
    mlp_relu_kernel<<<NB, 128, 0, stream>>>(h,  l1w, l1b, h1, 256);
    mlp_relu_kernel<<<NB, 64,  0, stream>>>(h1, l2w, l2b, h2, 128);
    head_kernel<<<1, NB, 0, stream>>>(h2, l3w, l3b, out);
}
